// convolution_75196287418639
// MI455X (gfx1250) — matrix operand order repaired and checked by reading, not compiled, not run on hardware
//
#include <hip/hip_runtime.h>

typedef __bf16 v16bf __attribute__((ext_vector_type(16)));
typedef float  v8f   __attribute__((ext_vector_type(8)));

// Problem constants (e3nn-style tensor product)
#define FEAT       40      // MUL0 + 3*MUL1 = 16 + 24
#define HD         64
#define WN         576     // 256 (w1) + 128 (w2) + 64 (w3) + 128 (w4)
#define WAVES      8
#define EPW        16      // edges per wave (WMMA M)
#define EB_STRIDE  72      // per-edge staging floats (padded)

__device__ __constant__ float INV_SQRT3 = 0.5773502691896257f;
__device__ __constant__ float PW0       = 0.20412414523193150f;  // sqrt(1/24)
__device__ __constant__ float PW1       = 0.35355339059327373f;  // sqrt(3/24)
__device__ __constant__ float SILU_N    = 1.6791767923989418f;
__device__ __constant__ float OUT_SCALE = 0.25f;                 // 1/sqrt(NUM_NEIGHBORS)

// ---------------------------------------------------------------------------
// Prep: convert W1 (64x64) and W2 (64x576) f32 -> bf16, laid out in WMMA
// B-fragment order.  B frag (32x16, bf16): lane l holds column n = 16*t + l%16,
// element j holds K = 32*f + 16*(l/16) + j.  One frag = 32 lanes * 16 bf16.
// ws layout: [0, 4096) = W1 frags (4 tiles x 2 kfrags), [4096, 40960) = W2.
// ---------------------------------------------------------------------------
__global__ void prep_weights(const float* __restrict__ W1,
                             const float* __restrict__ W2,
                             __bf16* __restrict__ ws) {
  int tid = blockIdx.x * blockDim.x + threadIdx.x;
  if (tid < 4096) {
    int j = tid & 15, lane = (tid >> 4) & 31, f = (tid >> 9) & 1, t = tid >> 10;
    int k = 32 * f + 16 * (lane >> 4) + j;
    int n = 16 * t + (lane & 15);
    ws[tid] = (__bf16)W1[k * HD + n];
  } else if (tid < 4096 + 36864) {
    int r = tid - 4096;
    int j = r & 15, lane = (r >> 4) & 31, f = (r >> 9) & 1, t = r >> 10; // t 0..35
    int k = 32 * f + 16 * (lane >> 4) + j;
    int n = 16 * t + (lane & 15);
    ws[tid] = (__bf16)W2[k * WN + n];
  }
}

// ---------------------------------------------------------------------------
// Fused: radial MLP (bf16 WMMA) + tensor product + segment-sum scatter.
// One wave per 16-edge tile; 8 tiles per block.
// ---------------------------------------------------------------------------
union F16 { v16bf v; int4 i[2]; };

__global__ __launch_bounds__(256) void edge_kernel(
    const float* __restrict__ node_input,
    const int*   __restrict__ edge_src,
    const int*   __restrict__ edge_dst,
    const float* __restrict__ edge_attr,
    const float* __restrict__ dist,
    const __bf16* __restrict__ wfrag,
    float* __restrict__ out,
    int n_edges) {

  __shared__ float eb[WAVES][EPW][EB_STRIDE];                 // per-edge scalars
  __shared__ __align__(16) __bf16 hb[WAVES][EPW][HD];         // hidden activations

  const int  lane = threadIdx.x & 31;
  const int  wave = threadIdx.x >> 5;
  const int  m    = lane & 15;            // WMMA row / column index
  const int  half = lane >> 4;
  const long e0   = ((long)blockIdx.x * WAVES + wave) * EPW;

  // ---- Phase A: lanes 0..15 gather + precompute per-edge scalars into LDS
  if (half == 0) {
    long e = e0 + m;
    bool valid = e < n_edges;
    long ec = valid ? e : (long)(n_edges - 1);
    const float* x = node_input + (long)edge_dst[ec] * FEAT;
    const float* y = edge_attr + ec * 4;
    float y0 = y[0], y1a = y[1], y1b = y[2], y1c = y[3];
    float vs = valid ? 1.0f : 0.0f;
    float* row = eb[wave][m];
#pragma unroll
    for (int u = 0; u < 16; ++u) {        // a0[u] = x0*y0 ; x0[u]
      float x0 = x[u];
      row[u]      = vs * x0 * y0;
      row[24 + u] = vs * x0;
    }
#pragma unroll
    for (int u = 0; u < 8; ++u) {         // a1[u] = x1.y1 ; x1*y0
      float xa = x[16 + 3 * u], xb = x[17 + 3 * u], xc = x[18 + 3 * u];
      row[16 + u]       = vs * (xa * y1a + xb * y1b + xc * y1c);
      row[40 + 3 * u]     = vs * xa * y0;
      row[40 + 3 * u + 1] = vs * xb * y0;
      row[40 + 3 * u + 2] = vs * xc * y0;
    }
    row[64] = y1a; row[65] = y1b; row[66] = y1c;
    row[67] = __int_as_float(edge_src[ec]);
  }

  // ---- Phase A': A fragments of dist_embedding (contiguous K order, same as the B frags: lane m,
  //      element j -> K = j + 16*half; two contiguous runs of 8)
  long eA = e0 + m; if (eA >= n_edges) eA = n_edges - 1;
  const float* drow = dist + eA * HD;
  F16 a1f[2];
#pragma unroll
  for (int f = 0; f < 2; ++f) {
#pragma unroll
    for (int run = 0; run < 2; ++run) {
      int k0 = 16 * half + 32 * f + 8 * run;
      float4 p0 = *reinterpret_cast<const float4*>(drow + k0);
      float4 p1 = *reinterpret_cast<const float4*>(drow + k0 + 4);
      int b = run * 8;
      a1f[f].v[b + 0] = (__bf16)p0.x; a1f[f].v[b + 1] = (__bf16)p0.y;
      a1f[f].v[b + 2] = (__bf16)p0.z; a1f[f].v[b + 3] = (__bf16)p0.w;
      a1f[f].v[b + 4] = (__bf16)p1.x; a1f[f].v[b + 5] = (__bf16)p1.y;
      a1f[f].v[b + 6] = (__bf16)p1.z; a1f[f].v[b + 7] = (__bf16)p1.w;
    }
  }

  __syncthreads();

  // ---- Phase B: MLP layer 1 (16x64 @ 64x64), SiLU, stage H (bf16) in LDS.
  //      Fold the two 1/sqrt(64) factors: store (SILU_N/8) * silu(c/8).
#pragma unroll
  for (int t = 0; t < 4; ++t) {
    v8f c = {};
#pragma unroll
    for (int f = 0; f < 2; ++f) {
      F16 b;
      const int4* bp = reinterpret_cast<const int4*>(wfrag) +
                       (long)((t * 2 + f) * 32 + lane) * 2;
      b.i[0] = bp[0]; b.i[1] = bp[1];
      c = __builtin_amdgcn_wmma_f32_16x16x32_bf16(false, a1f[f].v,
                                                  false, b.v,
                                                  (short)0, c, false, false);
    }
#pragma unroll
    for (int r = 0; r < 8; ++r) {         // C layout: M = r + 8*half, N = m
      float v = 0.125f * c[r];
      float h = (0.125f * SILU_N) * (v / (1.0f + __expf(-v)));
      hb[wave][r + 8 * half][16 * t + m] = (__bf16)h;
    }
  }

  __syncthreads();

  // ---- Phase C: reload H as A fragments (contiguous 16B runs -> ds_load_b128)
  F16 a2f[2];
  {
    const __bf16* hrow = hb[wave][m];
#pragma unroll
    for (int f = 0; f < 2; ++f)
#pragma unroll
      for (int run = 0; run < 2; ++run)
        a2f[f].i[run] = *reinterpret_cast<const int4*>(
            hrow + 16 * half + 32 * f + 8 * run);
  }

  // ---- Phase D: MLP layer 2 (16x64 @ 64x576) streamed tile-by-tile and
  //      consumed immediately by the tensor product contraction over u.
  const float (*ebw)[EB_STRIDE] = (const float (*)[EB_STRIDE])eb[wave];
  const int4* w2base = reinterpret_cast<const int4*>(wfrag + 4096);

  auto mma_tile = [&](int t) -> v8f {
    v8f c = {};
#pragma unroll
    for (int f = 0; f < 2; ++f) {
      F16 b;
      const int4* bp = w2base + (long)((t * 2 + f) * 32 + lane) * 2;
      b.i[0] = bp[0]; b.i[1] = bp[1];
      c = __builtin_amdgcn_wmma_f32_16x16x32_bf16(false, a2f[f].v,
                                                  false, b.v,
                                                  (short)0, c, false, false);
    }
    return c;
  };

  v8f b1 = {}, b2 = {}, b4 = {};
  v8f b3a = {}, b3b = {}, b3c = {};
  const int up = (m >> 3);                 // u parity select for w2/w3 regions

  for (int t = 0; t < 16; ++t) {           // w1: c = u*16 + w -> u = t, w = m
    v8f c = mma_tile(t);
#pragma unroll
    for (int r = 0; r < 8; ++r) b1[r] += ebw[r + 8 * half][t] * c[r];
  }
  for (int tt = 0; tt < 8; ++tt) {         // w2: u = 2*tt + up, w = m&7
    v8f c = mma_tile(16 + tt);
    int u = 2 * tt + up;
#pragma unroll
    for (int r = 0; r < 8; ++r) b2[r] += ebw[r + 8 * half][24 + u] * c[r];
  }
  for (int tt = 0; tt < 4; ++tt) {         // w3: u = 2*tt + up, w = m&7
    v8f c = mma_tile(24 + tt);
    int u = 2 * tt + up;
#pragma unroll
    for (int r = 0; r < 8; ++r) {
      const float* p = &ebw[r + 8 * half][40 + 3 * u];
      b3a[r] += p[0] * c[r];
      b3b[r] += p[1] * c[r];
      b3c[r] += p[2] * c[r];
    }
  }
  for (int tt = 0; tt < 8; ++tt) {         // w4: u = tt, w = m
    v8f c = mma_tile(28 + tt);
#pragma unroll
    for (int r = 0; r < 8; ++r) b4[r] += ebw[r + 8 * half][16 + tt] * c[r];
  }

  // ---- Phase E: epilogue + scatter (non-returning global_atomic_add_f32).
  //      out0 owned uniquely per (lane,r); out1 partial sums from lane pairs
  //      (col w, col w+8) both atomically added — associative, no shuffle.
  const int w8 = m & 7;
  const float c1 = OUT_SCALE * PW1 * INV_SQRT3;
#pragma unroll
  for (int r = 0; r < 8; ++r) {
    int me  = r + 8 * half;
    int src = __float_as_int(ebw[me][67]);
    float* op = out + (long)src * FEAT;
    atomicAdd(op + m, OUT_SCALE * PW0 * (b1[r] + INV_SQRT3 * b4[r]));
    float y1a = ebw[me][64], y1b = ebw[me][65], y1c = ebw[me][66];
    atomicAdd(op + 16 + 3 * w8 + 0, c1 * (b2[r] * y1a + b3a[r]));
    atomicAdd(op + 16 + 3 * w8 + 1, c1 * (b2[r] * y1b + b3b[r]));
    atomicAdd(op + 16 + 3 * w8 + 2, c1 * (b2[r] * y1c + b3c[r]));
  }
}

// ---------------------------------------------------------------------------
extern "C" void kernel_launch(void* const* d_in, const int* in_sizes, int n_in,
                              void* d_out, int out_size, void* d_ws, size_t ws_size,
                              hipStream_t stream) {
  const float* node_input = (const float*)d_in[0];
  const int*   edge_src   = (const int*)d_in[1];
  const int*   edge_dst   = (const int*)d_in[2];
  const float* edge_attr  = (const float*)d_in[3];
  const float* dist       = (const float*)d_in[4];
  const float* W1         = (const float*)d_in[5];
  const float* W2         = (const float*)d_in[6];
  float*  out = (float*)d_out;
  __bf16* ws  = (__bf16*)d_ws;

  const int n_edges = in_sizes[1];

  hipMemsetAsync(d_out, 0, (size_t)out_size * sizeof(float), stream);
  prep_weights<<<(40960 + 255) / 256, 256, 0, stream>>>(W1, W2, ws);

  int tiles  = (n_edges + EPW - 1) / EPW;       // 25000 for E = 400000
  int blocks = (tiles + WAVES - 1) / WAVES;     // 3125
  edge_kernel<<<blocks, 256, 0, stream>>>(node_input, edge_src, edge_dst,
                                          edge_attr, dist, ws, out, n_edges);
}
